// ST_60997125538445
// MI455X (gfx1250) — matrix operand order repaired and checked by reading, not compiled, not run on hardware
//
#include <hip/hip_runtime.h>
#include <hip/hip_bf16.h>
#include <stdint.h>

// ---------------- problem constants ----------------
#define NWAY   5
#define KSHOT  5
#define PROJ_  160
#define FEAT_  640
#define BQ_    300     // b*q = 4*75
#define MQ_    100     // h*w
#define MQP_   112     // M_q padded to 7*16
#define MS_    500     // K_SHOT*h*w
#define MSP_   512     // padded to 32*16

typedef __bf16 bf16_t;
typedef __attribute__((ext_vector_type(16))) __bf16 bf16x16;
typedef __attribute__((ext_vector_type(8)))  __bf16 bf16x8;
typedef __attribute__((ext_vector_type(8)))  float  floatx8;

// ---------------- helpers ----------------
__device__ __forceinline__ unsigned int ordf(float f) {
  unsigned int u = __float_as_uint(f);
  return (u & 0x80000000u) ? ~u : (u | 0x80000000u);
}
__device__ __forceinline__ float deordf(unsigned int o) {
  unsigned int u = (o & 0x80000000u) ? (o & 0x7FFFFFFFu) : ~o;
  return __uint_as_float(u);
}

__device__ __forceinline__ floatx8 wmma_bf16(bf16x16 a, bf16x16 b, floatx8 c) {
  // D = A(16x32) * B(32x16) + C, fp32 accumulate
  return __builtin_amdgcn_wmma_f32_16x16x32_bf16(false, a, false, b, (short)0, c, false, false);
}

// Async global->LDS 16-byte copy (CDNA5 GLOBAL_LOAD_ASYNC_TO_LDS_B128,
// tracked by ASYNCcnt; bypasses VGPRs). LDS byte address = low 32 bits of
// the generic pointer (LDS aperture: addr[31:0] = LDS offset).
__device__ __forceinline__ void async_g2l_b128(void* lds_dst, const void* gsrc) {
  unsigned int lds = (unsigned int)(uintptr_t)lds_dst;
  asm volatile("global_load_async_to_lds_b128 %0, %1, off"
               :: "v"(lds), "v"(gsrc) : "memory");
}
__device__ __forceinline__ void async_wait0() {
  asm volatile("s_wait_asynccnt 0x0" ::: "memory");
}

// A-fragment (16x32 bf16) from row-major [M][K] bf16 (LDS or global), ld in elems.
// Lane L: M = L&15; elements 0..7 -> K = 8*(L>>4)+e ; 8..15 -> K = 16+8*(L>>4)+e
__device__ __forceinline__ bf16x16 load_a_bf16(const bf16_t* base, int ld, int lane) {
  int m = lane & 15;
  int h = (lane >> 4) << 3;
  const bf16_t* r = base + m * ld;
  bf16x8 lo = *reinterpret_cast<const bf16x8*>(r + h);
  bf16x8 hi = *reinterpret_cast<const bf16x8*>(r + 16 + h);
  bf16x16 a;
#pragma unroll
  for (int e = 0; e < 8; ++e) { a[e] = lo[e]; a[8 + e] = hi[e]; }
  return a;
}

// B-fragment (32x16 bf16) from [K][N] bf16: lane L holds column N=L&15;
// elements 0..7 -> K = 8*(L>>4)+e ; 8..15 -> K = 16+8*(L>>4)+e
__device__ __forceinline__ bf16x16 load_b_bf16(const bf16_t* base, long long ld, int lane) {
  const bf16_t* c = base + (lane & 15);
  int h = (lane >> 4) << 3;
  bf16x16 b;
#pragma unroll
  for (int e = 0; e < 8; ++e) { b[e] = c[(long long)(h + e) * ld]; b[8 + e] = c[(long long)(16 + h + e) * ld]; }
  return b;
}

// ---------------- smem layout sizes ----------------
#define SM_P1  (640 * 120 * 2 + MQP_ * 4)                         // 154048
#define SM_S1  (MQP_ * 8 + NWAY * MQP_ * 4 + MQP_ * 168 * 2)      // 40768
#define S2_E_BYTES   (MQP_ * 520 * 2)                             // 116480
#define S2_AL_BYTES  (MQP_ * 136 * 2)                             // 30464
#define S2_QF_BYTES  (MQP_ * 168 * 2)                             // 37632
#define SM_S2  (S2_E_BYTES + S2_AL_BYTES + S2_QF_BYTES + 4*MQP_*4 + MQP_*4) // 186816

// =====================================================================
// K0: staging — convert all WMMA operands to zero-padded bf16 once.
// =====================================================================
__global__ void __launch_bounds__(256) k_stage_w(
    const float* __restrict__ Wq, const float* __restrict__ Wk,
    const float* __restrict__ Wv,
    bf16_t* __restrict__ WqB, bf16_t* __restrict__ WkB, bf16_t* __restrict__ WvB)
{
  int idx = blockIdx.x * 256 + threadIdx.x;   // 614400 total
  if (idx < 102400)            WqB[idx]          = (bf16_t)Wq[idx];
  else if (idx < 204800)       WkB[idx - 102400] = (bf16_t)Wk[idx - 102400];
  else                         WvB[idx - 204800] = (bf16_t)Wv[idx - 204800];
}

__global__ void __launch_bounds__(256) k_stage_xq(
    const float* __restrict__ query_xf,   // [300][640][100]
    bf16_t* __restrict__ xqB)             // [300][640][112], cols >=100 zero
{
  int idx = blockIdx.x * 256 + threadIdx.x;       // 300*640*112 = 21,504,000
  int i  = idx % MQP_;
  int rc = idx / MQP_;                            // bq*640 + c
  float v = (i < MQ_) ? query_xf[(long long)rc * MQ_ + i] : 0.f;
  xqB[idx] = (bf16_t)v;
}

__global__ void __launch_bounds__(256) k_stage_xs(
    const float* __restrict__ support_xf, // [4][25][640][100]
    bf16_t* __restrict__ xsB)             // [20][640][512], j = shot*100+m, >=500 zero
{
  int idx = blockIdx.x * 256 + threadIdx.x;       // 20*640*512 = 6,553,600
  int j  = idx % MSP_;
  int gc = idx / MSP_;                            // g*640 + c
  int c  = gc % FEAT_;
  int g  = gc / FEAT_;                            // b*5 + n
  float v = 0.f;
  if (j < MS_) {
    int shot = j / MQ_, m = j % MQ_;
    v = support_xf[(((long long)g * KSHOT + shot) * FEAT_ + c) * MQ_ + m];
  }
  xsB[idx] = (bf16_t)v;
}

// =====================================================================
// P1: query projections.  qfT[bq][112][160] = (Wq @ xq)^T * 1/sqrt(PROJ)
//     vqn[bq][640][112]  = l2norm_channels(Wv @ xq)
// =====================================================================
__global__ void __launch_bounds__(256) k_proj_query(
    const bf16_t* __restrict__ xqB,       // [300][640][112]
    const bf16_t* __restrict__ WqB,       // [160][640]
    const bf16_t* __restrict__ WvB,       // [640][640]
    bf16_t* __restrict__ qfT,             // [300][112][160]
    bf16_t* __restrict__ vqn)             // [300][640][112]
{
  extern __shared__ char smem[];
  bf16_t* vqs   = (bf16_t*)smem;                   // [640][120] bf16
  float*  sumsq = (float*)(smem + 640 * 120 * 2);  // [112]

  const int bq   = blockIdx.x;
  const int tid  = threadIdx.x, lane = tid & 31, wv = tid >> 5;
  const bf16_t* xq = xqB + (long long)bq * (FEAT_ * MQP_);
  const float scale = 0.07905694150420949f;  // 1/sqrt(160)

  // 350 tiles: [0,70) = qf (10 x 7), [70,350) = vq (40 x 7)
  for (int t = wv; t < 350; t += 8) {
    bool isQ = (t < 70);
    int tt = isQ ? t : t - 70;
    int mt = tt / 7, nt = tt % 7;
    const bf16_t* W = isQ ? WqB : WvB;
    floatx8 acc = {};
    for (int k0 = 0; k0 < FEAT_; k0 += 32) {
      bf16x16 a = load_a_bf16(W + (mt * 16) * FEAT_ + k0, FEAT_, lane);
      bf16x16 b = load_b_bf16(xq + (long long)k0 * MQP_ + nt * 16, MQP_, lane);
      acc = wmma_bf16(a, b, acc);
    }
    int i    = nt * 16 + (lane & 15);
    int half = lane >> 4;
    if (isQ) {
#pragma unroll
      for (int r = 0; r < 8; ++r) {
        int p = mt * 16 + r + 8 * half;
        qfT[((long long)bq * MQP_ + i) * PROJ_ + p] = (bf16_t)(acc[r] * scale);
      }
    } else {
#pragma unroll
      for (int r = 0; r < 8; ++r) {
        int o = mt * 16 + r + 8 * half;
        vqs[o * 120 + i] = (bf16_t)acc[r];
      }
    }
  }
  __syncthreads();
  // deterministic per-column sumsq over 640 channels -> reciprocal norm
  if (tid < MQP_) {
    float s = 0.f;
    for (int o = 0; o < FEAT_; ++o) {
      float v = (float)vqs[o * 120 + tid];
      s += v * v;
    }
    sumsq[tid] = 1.0f / fmaxf(sqrtf(s), 1e-12f);
  }
  __syncthreads();
  for (int idx = tid; idx < FEAT_ * MQP_; idx += 256) {
    int o = idx / MQP_, i = idx % MQP_;
    vqn[((long long)bq * FEAT_ + o) * MQP_ + i] = (bf16_t)((float)vqs[o * 120 + i] * sumsq[i]);
  }
}

// =====================================================================
// P2: support projections per (b,n).
//   skB[g][160][512] = Wk @ xs      (B-operand layout [K=p][N=j])
//   svT[g][512][640] = (Wv @ xs)^T  (B-operand layout [K=j][N=o])
// =====================================================================
__global__ void __launch_bounds__(256) k_proj_support(
    const bf16_t* __restrict__ xsB,        // [20][640][512]
    const bf16_t* __restrict__ WkB,
    const bf16_t* __restrict__ WvB,
    bf16_t* __restrict__ skB,              // [20][160][512]
    bf16_t* __restrict__ svT)              // [20][512][640]
{
  const int g = blockIdx.x;          // b*5 + n
  const int tid = threadIdx.x, lane = tid & 31, wv = tid >> 5;
  const bf16_t* xs = xsB + (long long)g * (FEAT_ * MSP_);

  // tiles: sk 10x32 = 320, sv 40x32 = 1280 -> 1600
  for (int t = wv; t < 1600; t += 8) {
    bool isK = (t < 320);
    int tt = isK ? t : t - 320;
    int mt = tt / 32, nt = tt % 32;
    const bf16_t* W = isK ? WkB : WvB;
    floatx8 acc = {};
    for (int k0 = 0; k0 < FEAT_; k0 += 32) {
      bf16x16 a = load_a_bf16(W + (mt * 16) * FEAT_ + k0, FEAT_, lane);
      bf16x16 bv = load_b_bf16(xs + (long long)k0 * MSP_ + nt * 16, MSP_, lane);
      acc = wmma_bf16(a, bv, acc);
    }
    int jn   = nt * 16 + (lane & 15);
    int half = lane >> 4;
    if (isK) {
#pragma unroll
      for (int r = 0; r < 8; ++r) {
        int p = mt * 16 + r + 8 * half;
        skB[((long long)g * PROJ_ + p) * MSP_ + jn] = (bf16_t)acc[r];
      }
    } else {
#pragma unroll
      for (int r = 0; r < 8; ++r) {
        int o = mt * 16 + r + 8 * half;
        svT[((long long)g * MSP_ + jn) * FEAT_ + o] = (bf16_t)acc[r];
      }
    }
  }
}

// Async-stage qfT[bq] ([112][160] bf16, contiguous) into LDS with stride 168.
// 2240 x 16B transfers, direct global->LDS (no VGPR staging).
__device__ __forceinline__ void stage_qfs_async(bf16_t* qfs, const bf16_t* qsrc, int tid) {
  for (int idx = tid; idx < MQP_ * 20; idx += 256) {
    int row = idx / 20, c16 = idx % 20;       // 20 x 8 bf16 chunks per row
    async_g2l_b128(qfs + row * 168 + c16 * 8, qsrc + row * PROJ_ + c16 * 8);
  }
  async_wait0();
}

// =====================================================================
// S1: per (b,q): full simi pass, track per-row (max,argmax) over (n,j)
//     (packed u64 ds_max), and per-(n,i) row max (softmax shift).
//     Register-blocked: each wave owns a j-strip, B loaded once per K-step,
//     fed into 7 WMMAs (one per i row-tile).
// =====================================================================
__global__ void __launch_bounds__(256) k_mask(
    const bf16_t* __restrict__ qfT,   // [300][112][160]
    const bf16_t* __restrict__ skB,   // [20][160][512]
    float* __restrict__ maskG,        // [300][112]
    float* __restrict__ rmaxG)        // [300][5][112]
{
  extern __shared__ char smem[];
  unsigned long long* packed = (unsigned long long*)smem;                    // [112]
  unsigned int* rowmax = (unsigned int*)(smem + MQP_ * 8);                   // [5][112]
  bf16_t* qfs = (bf16_t*)(smem + MQP_ * 8 + NWAY * MQP_ * 4);                // [112][168]

  const int bq  = blockIdx.x;
  const int b   = bq / 75;
  const int tid = threadIdx.x, lane = tid & 31, wv = tid >> 5;

  stage_qfs_async(qfs, qfT + (long long)bq * MQP_ * PROJ_, tid);
  if (tid < MQP_) packed[tid] = 0ull;
  for (int idx = tid; idx < NWAY * MQP_; idx += 256) rowmax[idx] = 0u;
  __syncthreads();

  const int half = lane >> 4;
  for (int n = 0; n < NWAY; ++n) {
    const bf16_t* bb = skB + (long long)(b * 5 + n) * PROJ_ * MSP_;
    for (int js = 0; js < 4; ++js) {
      const int jt = wv + js * 8;                 // 8 waves x 4 -> all 32 strips
      floatx8 acc[7] = {};
      for (int k0 = 0; k0 < PROJ_; k0 += 32) {
        bf16x16 bv = load_b_bf16(bb + (long long)k0 * MSP_ + jt * 16, MSP_, lane);
#pragma unroll
        for (int it = 0; it < 7; ++it) {
          bf16x16 a = load_a_bf16(qfs + it * 16 * 168 + k0, 168, lane);
          acc[it] = wmma_bf16(a, bv, acc[it]);
        }
      }
      int j = jt * 16 + (lane & 15);
      if (j < MS_) {
        unsigned int lowk = 0xFFFFFFFFu - (unsigned int)(n * MS_ + j);
#pragma unroll
        for (int it = 0; it < 7; ++it) {
#pragma unroll
          for (int r = 0; r < 8; ++r) {
            int i = it * 16 + r + 8 * half;
            if (i < MQ_) {
              unsigned int ov = ordf(acc[it][r]);
              unsigned long long key = ((unsigned long long)ov << 32) | (unsigned long long)lowk;
              atomicMax(&packed[i], key);
              atomicMax(&rowmax[n * MQP_ + i], ov);
            }
          }
        }
      }
    }
  }
  __syncthreads();

  if (tid < MQP_) {
    float mk = 0.f;
    if (tid < MQ_) {
      unsigned long long mine = packed[tid];
      unsigned int mycol = 0xFFFFFFFFu - (unsigned int)(mine & 0xFFFFFFFFu);
      // winner of class_m column `mycol`: argmax_i {qn[i]==col ? max_i+1 : 0}, ties->lowest i
      int besti = 0;
      float bestv = -3.4e38f;
      for (int i2 = 0; i2 < MQ_; ++i2) {
        unsigned long long pk = packed[i2];
        unsigned int col2 = 0xFFFFFFFFu - (unsigned int)(pk & 0xFFFFFFFFu);
        float v = (col2 == mycol) ? (deordf((unsigned int)(pk >> 32)) + 1.0f) : 0.0f;
        if (v > bestv) { bestv = v; besti = i2; }
      }
      mk = (besti == tid) ? 1.f : 0.f;
    }
    maskG[(long long)bq * MQP_ + tid] = mk;
    for (int n = 0; n < NWAY; ++n)
      rmaxG[((long long)bq * NWAY + n) * MQP_ + tid] = deordf(rowmax[n * MQP_ + tid]);
  }
}

// =====================================================================
// S2: per (b,q,n): fused exp(simi)->LDS, softmax-normalized attention,
//     aligned = E @ svT (o-chunked), L2 row norms, simi2 = aligned @ vqn,
//     column-max + sum  ->  out[bq*5+n].  All GEMMs register-blocked 7x.
// =====================================================================
__global__ void __launch_bounds__(256) k_attend(
    const bf16_t* __restrict__ qfT,
    const bf16_t* __restrict__ skB,
    const bf16_t* __restrict__ svT,
    const bf16_t* __restrict__ vqn,
    const float* __restrict__ maskG,
    const float* __restrict__ rmaxG,
    float* __restrict__ out)
{
  extern __shared__ char smem[];
  bf16_t* E   = (bf16_t*)smem;                                   // [112][520]
  bf16_t* AL  = (bf16_t*)(smem + S2_E_BYTES);                    // [112][136]
  bf16_t* qfs = (bf16_t*)(smem + S2_E_BYTES + S2_AL_BYTES);      // [112][168]
  float* Zr   = (float*)(smem + S2_E_BYTES + S2_AL_BYTES + S2_QF_BYTES);
  float* ss   = Zr + MQP_;
  float* rmx  = ss + MQP_;
  float* msk  = rmx + MQP_;
  unsigned int* colmax = (unsigned int*)(msk + MQP_);

  const int g  = blockIdx.x;       // bq*5 + n
  const int bq = g / NWAY, n = g % NWAY;
  const int b  = bq / 75;
  const int sg = b * 5 + n;
  const int tid = threadIdx.x, lane = tid & 31, wv = tid >> 5;
  const int half = lane >> 4;

  stage_qfs_async(qfs, qfT + (long long)bq * MQP_ * PROJ_, tid);
  if (tid < MQP_) {
    msk[tid]    = maskG[(long long)bq * MQP_ + tid];
    rmx[tid]    = rmaxG[((long long)bq * NWAY + n) * MQP_ + tid];
    ss[tid]     = 0.f;
    colmax[tid] = 0u;
  }
  __syncthreads();

  // --- Phase 1: E = masked exp(simi - rowmax)  (zero-padded to [112][512])
  const bf16_t* skb = skB + (long long)sg * PROJ_ * MSP_;
  for (int js = 0; js < 4; ++js) {
    const int jt = wv + js * 8;
    floatx8 acc[7] = {};
    for (int k0 = 0; k0 < PROJ_; k0 += 32) {
      bf16x16 bv = load_b_bf16(skb + (long long)k0 * MSP_ + jt * 16, MSP_, lane);
#pragma unroll
      for (int it = 0; it < 7; ++it) {
        bf16x16 a = load_a_bf16(qfs + it * 16 * 168 + k0, 168, lane);
        acc[it] = wmma_bf16(a, bv, acc[it]);
      }
    }
    int j = jt * 16 + (lane & 15);
#pragma unroll
    for (int it = 0; it < 7; ++it) {
#pragma unroll
      for (int r = 0; r < 8; ++r) {
        int i = it * 16 + r + 8 * half;
        float e = 0.f;
        if (i < MQ_ && j < MS_)
          e = (msk[i] != 0.f) ? __expf(acc[it][r] - rmx[i]) : 1.0f;  // masked row -> uniform
        E[i * 520 + j] = (bf16_t)e;
      }
    }
  }
  __syncthreads();

  // softmax denominators (deterministic per-row)
  if (tid < MQP_) {
    float s = 0.f;
    for (int j = 0; j < MS_; ++j) s += (float)E[tid * 520 + j];
    Zr[tid] = (s > 0.f) ? (1.0f / s) : 0.f;
  }
  __syncthreads();

  // --- Phases 2/3: o-chunked aligned GEMM + accumulated simi2 GEMM
  floatx8 acc2[7] = {};                     // wave wv(<7) owns m-strip mt = wv
  const bf16_t* svb = svT + (long long)sg * MSP_ * FEAT_;
  const bf16_t* vqb = vqn + (long long)bq * FEAT_ * MQP_;

  for (int oc = 0; oc < FEAT_; oc += 128) {
    // aligned chunk [112][128] = (E * diag(1/Z)) @ svT[:, oc:oc+128]; wave owns ot = wv
    {
      floatx8 accA[7] = {};
      for (int k0 = 0; k0 < MSP_; k0 += 32) {
        bf16x16 bv = load_b_bf16(svb + (long long)k0 * FEAT_ + oc + wv * 16, FEAT_, lane);
#pragma unroll
        for (int it = 0; it < 7; ++it) {
          bf16x16 a = load_a_bf16(E + it * 16 * 520 + k0, 520, lane);
          accA[it] = wmma_bf16(a, bv, accA[it]);
        }
      }
      int ol = wv * 16 + (lane & 15);
#pragma unroll
      for (int it = 0; it < 7; ++it) {
#pragma unroll
        for (int r = 0; r < 8; ++r) {
          int i = it * 16 + r + 8 * half;
          AL[i * 136 + ol] = (bf16_t)(accA[it][r] * Zr[i]);
        }
      }
    }
    __syncthreads();
    // accumulate row sum-of-squares (single owner per row -> deterministic)
    if (tid < MQP_) {
      float s = 0.f;
      for (int o = 0; o < 128; ++o) {
        float v = (float)AL[tid * 136 + o];
        s += v * v;
      }
      ss[tid] += s;
    }
    // simi2 partial: acc2 += AL @ vqn[oc:oc+128, :]; wave wv<7 owns mt = wv
    if (wv < 7) {
      for (int k0 = 0; k0 < 128; k0 += 32) {
        bf16x16 bv = load_b_bf16(vqb + (long long)(oc + k0) * MQP_ + wv * 16, MQP_, lane);
#pragma unroll
        for (int it = 0; it < 7; ++it) {
          bf16x16 a = load_a_bf16(AL + it * 16 * 136 + k0, 136, lane);
          acc2[it] = wmma_bf16(a, bv, acc2[it]);
        }
      }
    }
    __syncthreads();
  }

  // --- reciprocal L2 norms, then column max of simi2 over rows i
  if (tid < MQP_) ss[tid] = 1.0f / fmaxf(sqrtf(ss[tid]), 1e-12f);
  __syncthreads();
  if (wv < 7) {
    int m = wv * 16 + (lane & 15);
    if (m < MQ_) {
#pragma unroll
      for (int it = 0; it < 7; ++it) {
#pragma unroll
        for (int r = 0; r < 8; ++r) {
          int i = it * 16 + r + 8 * half;
          if (i < MQ_) {
            float v = acc2[it][r] * ss[i];
            atomicMax(&colmax[m], ordf(v));
          }
        }
      }
    }
  }
  __syncthreads();
  if (tid == 0) {
    float s = 0.f;
    for (int m = 0; m < MQ_; ++m) s += deordf(colmax[m]);
    out[g] = s;
  }
}

// =====================================================================
extern "C" void kernel_launch(void* const* d_in, const int* in_sizes, int n_in,
                              void* d_out, int out_size, void* d_ws, size_t ws_size,
                              hipStream_t stream) {
  (void)in_sizes; (void)n_in; (void)out_size; (void)ws_size;
  const float* support_xf = (const float*)d_in[0];
  const float* query_xf   = (const float*)d_in[2];
  const float* Wq         = (const float*)d_in[4];
  const float* Wk         = (const float*)d_in[5];
  const float* Wv         = (const float*)d_in[6];
  float* out = (float*)d_out;

  char* ws = (char*)d_ws;
  size_t o_qfT  = 0;
  size_t o_vqn  = o_qfT  + (size_t)BQ_ * MQP_ * PROJ_ * 2;   // 10,752,000
  size_t o_skB  = o_vqn  + (size_t)BQ_ * FEAT_ * MQP_ * 2;   // 43,008,000
  size_t o_svT  = o_skB  + (size_t)20 * PROJ_ * MSP_ * 2;    //  3,276,800
  size_t o_mask = o_svT  + (size_t)20 * MSP_ * FEAT_ * 2;    // 13,107,200
  size_t o_rmax = o_mask + (size_t)BQ_ * MQP_ * 4;           //    134,400
  size_t o_WqB  = o_rmax + (size_t)BQ_ * NWAY * MQP_ * 4;    //    672,000
  size_t o_WkB  = o_WqB  + (size_t)PROJ_ * FEAT_ * 2;        //    204,800
  size_t o_WvB  = o_WkB  + (size_t)PROJ_ * FEAT_ * 2;        //    204,800
  size_t o_xqB  = o_WvB  + (size_t)FEAT_ * FEAT_ * 2;        //    819,200
  size_t o_xsB  = o_xqB  + (size_t)BQ_ * FEAT_ * MQP_ * 2;   // 43,008,000

  bf16_t* qfT  = (bf16_t*)(ws + o_qfT);
  bf16_t* vqn  = (bf16_t*)(ws + o_vqn);
  bf16_t* skB  = (bf16_t*)(ws + o_skB);
  bf16_t* svT  = (bf16_t*)(ws + o_svT);
  float*  mask = (float*)(ws + o_mask);
  float*  rmax = (float*)(ws + o_rmax);
  bf16_t* WqB  = (bf16_t*)(ws + o_WqB);
  bf16_t* WkB  = (bf16_t*)(ws + o_WkB);
  bf16_t* WvB  = (bf16_t*)(ws + o_WvB);
  bf16_t* xqB  = (bf16_t*)(ws + o_xqB);
  bf16_t* xsB  = (bf16_t*)(ws + o_xsB);

  (void)hipFuncSetAttribute((const void*)k_proj_query, hipFuncAttributeMaxDynamicSharedMemorySize, SM_P1);
  (void)hipFuncSetAttribute((const void*)k_mask,       hipFuncAttributeMaxDynamicSharedMemorySize, SM_S1);
  (void)hipFuncSetAttribute((const void*)k_attend,     hipFuncAttributeMaxDynamicSharedMemorySize, SM_S2);

  // staging (pure elementwise bf16 conversion, zero-padded)
  k_stage_w <<<(102400*2 + 409600) / 256, 256, 0, stream>>>(Wq, Wk, Wv, WqB, WkB, WvB);
  k_stage_xq<<<(BQ_ * FEAT_ * MQP_) / 256, 256, 0, stream>>>(query_xf, xqB);
  k_stage_xs<<<(20 * FEAT_ * MSP_) / 256, 256, 0, stream>>>(support_xf, xsB);

  // projections
  k_proj_query  <<<BQ_, 256, SM_P1, stream>>>(xqB, WqB, WvB, qfT, vqn);
  k_proj_support<<<20,  256, 0,     stream>>>(xsB, WkB, WvB, skB, svT);

  // fused attention
  k_mask  <<<BQ_,        256, SM_S1, stream>>>(qfT, skB, mask, rmax);
  k_attend<<<BQ_ * NWAY, 256, SM_S2, stream>>>(qfT, skB, svT, vqn, mask, rmax, out);
}
